// GAPP_85907935855386
// MI455X (gfx1250) — matrix operand order repaired and checked by reading, not compiled, not run on hardware
//
#include <hip/hip_runtime.h>
#include <hip/hip_bf16.h>

// APPNP (MLP encoder + K personalized-PageRank propagations + log_softmax)
// for MI455X / gfx1250: bf16 WMMA encoder GEMMs (v_wmma_f32_16x16x32_bf16),
// L2-resident scatter/gather propagation, wave32 shfl log_softmax.

#define APPNP_ALPHA 0.1f
#define APPNP_K 5

typedef __attribute__((ext_vector_type(16))) __bf16 v16bf;
typedef __attribute__((ext_vector_type(8)))  __bf16 v8bf;
typedef __attribute__((ext_vector_type(8)))  float  v8f;
typedef __attribute__((ext_vector_type(4)))  float  v4f;

__device__ __forceinline__ __bf16 f2bf(float f) { return (__bf16)f; }

// ---------------------------------------------------------------------------
// GEMM1: h1[N,HID] = relu(x[N,IN] @ W1[HID,IN]^T + b1), stored bf16.
// Block = 256 threads = 8 waves; block owns 16 rows x 256 cols; each wave
// computes two 16x16 tiles via v_wmma_f32_16x16x32_bf16 over IN in steps of 32.
// Operand fragments loaded as float4 (global_load_b128) and cvt'd to bf16.
// ---------------------------------------------------------------------------
__global__ __launch_bounds__(256) void appnp_gemm1(
    const float* __restrict__ x, const float* __restrict__ W1,
    const float* __restrict__ b1, __bf16* __restrict__ h1,
    int N, int IN, int HID)
{
    const int lane = threadIdx.x & 31;
    const int wave = threadIdx.x >> 5;
    const int m    = lane & 15;
    const int half = lane >> 4;
    const int row0 = blockIdx.x * 16;

    int row = row0 + m;
    if (row >= N) row = N - 1;                 // clamp loads (uniform grid covers N)
    const float* arow = x + (size_t)row * IN;
    const bool fullTile = (row0 + 16 <= N);    // uniform: one scalar branch at stores

    for (int ct = 0; ct < 2; ++ct) {
        const int col0 = (wave * 2 + ct) * 16; // 0..240
        const int col  = col0 + m;             // B column per lane
        const float* brow = W1 + (size_t)col * IN;   // W1^T[k,col] = W1[col*IN+k]
        v8f acc = {};
        for (int kb = 0; kb < IN; kb += 32) {
            // A layout: e=0..7 -> K=kb+half*8+e ; e=8..15 -> K=kb+16+half*8+(e-8)
            const float* ap = arow + kb + half * 8;
            v4f a0 = *(const v4f*)(ap);
            v4f a1 = *(const v4f*)(ap + 4);
            v4f a2 = *(const v4f*)(ap + 16);
            v4f a3 = *(const v4f*)(ap + 20);
            // B layout: e=0..7 -> K=kb+half*8+e ; e=8..15 -> K=kb+16+half*8+(e-8)
            const float* bp = brow + kb + half * 8;
            v4f b0 = *(const v4f*)(bp);
            v4f b1v = *(const v4f*)(bp + 4);
            v4f b2 = *(const v4f*)(bp + 16);
            v4f b3 = *(const v4f*)(bp + 20);

            v16bf a, b;
            #pragma unroll
            for (int e = 0; e < 4; ++e) {
                a[e]      = f2bf(a0[e]);
                a[4 + e]  = f2bf(a1[e]);
                a[8 + e]  = f2bf(a2[e]);
                a[12 + e] = f2bf(a3[e]);
                b[e]      = f2bf(b0[e]);
                b[4 + e]  = f2bf(b1v[e]);
                b[8 + e]  = f2bf(b2[e]);
                b[12 + e] = f2bf(b3[e]);
            }
            acc = __builtin_amdgcn_wmma_f32_16x16x32_bf16(
                false, a, false, b, (short)0, acc, false, false);
        }
        const float bias = b1[col0 + m];
        __bf16* dcol = h1 + (size_t)(row0 + half * 8) * HID + col0 + m;
        if (fullTile) {                        // fast path: no exec-mask ladder
            #pragma unroll
            for (int v = 0; v < 8; ++v) {      // D: row = row0+half*8+v, col = col0+m
                float val = acc[v] + bias;
                dcol[(size_t)v * HID] = f2bf(val > 0.0f ? val : 0.0f);
            }
        } else {
            #pragma unroll
            for (int v = 0; v < 8; ++v) {
                if (row0 + half * 8 + v < N) {
                    float val = acc[v] + bias;
                    dcol[(size_t)v * HID] = f2bf(val > 0.0f ? val : 0.0f);
                }
            }
        }
    }
}

// ---------------------------------------------------------------------------
// GEMM2: h[N,OUT] = h1[N,HID](bf16) @ W2[OUT,HID]^T + b2, stored f32.
// Block = 128 threads = 4 waves; block owns 16 rows x 64 cols (1 tile/wave).
// A fragment: two 16-byte bf16 vector loads; B: four float4 loads + cvt.
// ---------------------------------------------------------------------------
__global__ __launch_bounds__(128) void appnp_gemm2(
    const __bf16* __restrict__ h1, const float* __restrict__ W2,
    const float* __restrict__ b2, float* __restrict__ h,
    int N, int HID, int OUT)
{
    const int lane = threadIdx.x & 31;
    const int wave = threadIdx.x >> 5;         // 0..3 -> column tile
    const int m    = lane & 15;
    const int half = lane >> 4;
    const int row0 = blockIdx.x * 16;

    int row = row0 + m;
    if (row >= N) row = N - 1;
    const __bf16* arow = h1 + (size_t)row * HID;
    const bool fullTile = (row0 + 16 <= N);

    const int col0 = wave * 16;
    const int col  = col0 + m;
    const float* brow = W2 + (size_t)col * HID;

    v8f acc = {};
    for (int kb = 0; kb < HID; kb += 32) {
        const __bf16* ap = arow + kb + half * 8;
        v8bf alo = *(const v8bf*)(ap);         // K = kb+half*8 .. +7
        v8bf ahi = *(const v8bf*)(ap + 16);    // K = kb+16+half*8 .. +7
        const float* bp = brow + kb + half * 8;
        v4f b0 = *(const v4f*)(bp);
        v4f b1v = *(const v4f*)(bp + 4);
        v4f b2 = *(const v4f*)(bp + 16);
        v4f b3 = *(const v4f*)(bp + 20);

        v16bf a, b;
        #pragma unroll
        for (int e = 0; e < 8; ++e) {
            a[e]     = alo[e];
            a[8 + e] = ahi[e];
        }
        #pragma unroll
        for (int e = 0; e < 4; ++e) {
            b[e]      = f2bf(b0[e]);
            b[4 + e]  = f2bf(b1v[e]);
            b[8 + e]  = f2bf(b2[e]);
            b[12 + e] = f2bf(b3[e]);
        }
        acc = __builtin_amdgcn_wmma_f32_16x16x32_bf16(
            false, a, false, b, (short)0, acc, false, false);
    }
    const float bias = b2[col0 + m];
    float* dcol = h + (size_t)(row0 + half * 8) * OUT + col0 + m;
    if (fullTile) {
        #pragma unroll
        for (int v = 0; v < 8; ++v) dcol[(size_t)v * OUT] = acc[v] + bias;
    } else {
        #pragma unroll
        for (int v = 0; v < 8; ++v)
            if (row0 + half * 8 + v < N) dcol[(size_t)v * OUT] = acc[v] + bias;
    }
}

// ---------------------------------------------------------------------------
// Degree (with self-loop) -> dinv = rsqrt(deg)
// ---------------------------------------------------------------------------
__global__ void appnp_deg_init(float* __restrict__ deg, int N) {
    int i = blockIdx.x * blockDim.x + threadIdx.x;
    if (i < N) deg[i] = 1.0f;                  // self-loop contribution
}

__global__ void appnp_deg_accum(const int* __restrict__ dst,
                                float* __restrict__ deg, int E) {
    int e = blockIdx.x * blockDim.x + threadIdx.x;
    if (e < E) atomicAdd(&deg[dst[e]], 1.0f);
}

__global__ void appnp_deg_inv(float* __restrict__ deg, int N) {
    int i = blockIdx.x * blockDim.x + threadIdx.x;
    if (i < N) {
        float d = deg[i];
        deg[i] = (d > 0.0f) ? rsqrtf(d) : 0.0f;
    }
}

// ---------------------------------------------------------------------------
// Propagation iteration:
//  init:  out = alpha*h + (1-alpha) * dinv[i]^2 * xk[i]          (self-loop)
//  edges: out[dst] += (1-alpha) * dinv[src]*dinv[dst] * xk[src]  (64 lanes/edge)
// ---------------------------------------------------------------------------
__global__ __launch_bounds__(256) void appnp_prop_init(
    const float* __restrict__ h, const float* __restrict__ xk,
    const float* __restrict__ dinv, float* __restrict__ out,
    int N, int OUT)
{
    size_t gid = (size_t)blockIdx.x * blockDim.x + threadIdx.x;
    if (gid >= (size_t)N * OUT) return;
    int row = (int)(gid / OUT);
    float di = dinv[row];
    out[gid] = APPNP_ALPHA * h[gid] + (1.0f - APPNP_ALPHA) * di * di * xk[gid];
}

__global__ __launch_bounds__(256) void appnp_prop_edges(
    const int* __restrict__ src, const int* __restrict__ dst,
    const float* __restrict__ dinv, const float* __restrict__ xk,
    float* __restrict__ out, int E)
{
    long long gid = (long long)blockIdx.x * blockDim.x + threadIdx.x;
    int e = (int)(gid >> 6);                   // OUT == 64 features per edge
    if (e >= E) return;
    int f = (int)(gid & 63);
    int s = src[e], d = dst[e];
    float w = (1.0f - APPNP_ALPHA) * dinv[s] * dinv[d];
    atomicAdd(out + (size_t)d * 64 + f, w * xk[(size_t)s * 64 + f]);
}

// ---------------------------------------------------------------------------
// Row-wise log_softmax over 64 features: one wave32 per row, 2 elems/lane.
// ---------------------------------------------------------------------------
__global__ __launch_bounds__(256) void appnp_logsoftmax(
    const float* __restrict__ xin, float* __restrict__ out, int N)
{
    const int lane = threadIdx.x & 31;
    const int wave = threadIdx.x >> 5;
    const int row  = blockIdx.x * 8 + wave;
    if (row >= N) return;
    const float* r = xin + (size_t)row * 64;
    float v0 = r[lane];
    float v1 = r[lane + 32];
    float mx = fmaxf(v0, v1);
    #pragma unroll
    for (int off = 16; off > 0; off >>= 1) mx = fmaxf(mx, __shfl_xor(mx, off, 32));
    float s = expf(v0 - mx) + expf(v1 - mx);
    #pragma unroll
    for (int off = 16; off > 0; off >>= 1) s += __shfl_xor(s, off, 32);
    float lse = mx + logf(s);
    out[(size_t)row * 64 + lane]      = v0 - lse;
    out[(size_t)row * 64 + lane + 32] = v1 - lse;
}

// ---------------------------------------------------------------------------
// Host-side orchestration (graph-capture safe: all launches on `stream`,
// workspace fully re-initialized by the kernels themselves each call).
// ---------------------------------------------------------------------------
extern "C" void kernel_launch(void* const* d_in, const int* in_sizes, int n_in,
                              void* d_out, int out_size, void* d_ws, size_t ws_size,
                              hipStream_t stream) {
    (void)n_in; (void)out_size; (void)ws_size;

    const float* x  = (const float*)d_in[0];
    const float* W1 = (const float*)d_in[1];
    const float* b1 = (const float*)d_in[2];
    const float* W2 = (const float*)d_in[3];
    const float* b2 = (const float*)d_in[4];
    const int*   ei = (const int*)d_in[5];     // int32 per harness convention

    const int HID = in_sizes[2];               // 256
    const int OUT = in_sizes[4];               // 64
    const int IN  = in_sizes[1] / HID;         // 512
    const int N   = in_sizes[0] / IN;          // 50000
    const int E   = in_sizes[5] / 2;           // 800000
    const int* src = ei;
    const int* dst = ei + E;

    // workspace carve-up (256B aligned): h1(bf16), h, xA, xB, dinv
    char*  ws  = (char*)d_ws;
    size_t off = 0;
    auto take = [&](size_t bytes) -> char* {
        char* p = ws + off;
        off += (bytes + 255) & ~(size_t)255;
        return p;
    };
    __bf16* h1   = (__bf16*)take((size_t)N * HID * sizeof(__bf16));
    float*  h    = (float*) take((size_t)N * OUT * sizeof(float));
    float*  xA   = (float*) take((size_t)N * OUT * sizeof(float));
    float*  xB   = (float*) take((size_t)N * OUT * sizeof(float));
    float*  dinv = (float*) take((size_t)N * sizeof(float));

    const int gTiles = (N + 15) / 16;          // 3125 (N%16==0)

    // Encoder
    appnp_gemm1<<<gTiles, 256, 0, stream>>>(x, W1, b1, h1, N, IN, HID);
    appnp_gemm2<<<gTiles, 128, 0, stream>>>(h1, W2, b2, h, N, HID, OUT);

    // gcn_norm with self-loops
    appnp_deg_init <<<(N + 255) / 256, 256, 0, stream>>>(dinv, N);
    appnp_deg_accum<<<(E + 255) / 256, 256, 0, stream>>>(dst, dinv, E);
    appnp_deg_inv  <<<(N + 255) / 256, 256, 0, stream>>>(dinv, N);

    // K iterations of PPR propagation (ping-pong xA/xB; x0 = h)
    const size_t feat  = (size_t)N * OUT;
    const int gFeat    = (int)((feat + 255) / 256);
    const long long ew = (long long)E * 64;
    const int gEdge    = (int)((ew + 255) / 256);

    const float* cur = h;
    float* bufs[2] = { xA, xB };
    for (int k = 0; k < APPNP_K; ++k) {
        float* nxt = bufs[k & 1];
        appnp_prop_init <<<gFeat, 256, 0, stream>>>(h, cur, dinv, nxt, N, OUT);
        appnp_prop_edges<<<gEdge, 256, 0, stream>>>(src, dst, dinv, cur, nxt, E);
        cur = nxt;
    }

    appnp_logsoftmax<<<(N + 7) / 8, 256, 0, stream>>>(cur, (float*)d_out, N);
}
